// MessagePassingNN_5059471475135
// MI455X (gfx1250) — compile-verified, matrix operand order checked by reading, not run on hardware
//
#include <hip/hip_runtime.h>
#include <hip/hip_bf16.h>

typedef _Float16 h16;
typedef __attribute__((ext_vector_type(16))) _Float16 v16h;
typedef __attribute__((ext_vector_type(8)))  _Float16 v8h;
typedef __attribute__((ext_vector_type(8)))  float    v8f;

#define GB 4      // batch
#define GN 256    // nodes
#define GF 64     // input features
#define GH 256    // hidden
#define GA 16     // actions
#define GT 3      // message passing steps
#define GM (GB*GN) // 1024 rows in all big GEMMs

// ---------------------------------------------------------------------------
// Fragment load: A is (M,K) f16 row-major, B is supplied as Wt (N,K) f16
// row-major (i.e. pre-transposed), so both fragments are two contiguous
// 16-byte chunks per lane, matching the CDNA5 16-bit A-matrix layout:
//   lanes 0-15:  halves 0..7 -> K = k0+0..7,   halves 8..15 -> K = k0+16..23
//   lanes 16-31: halves 0..7 -> K = k0+8..15,  halves 8..15 -> K = k0+24..31
// ---------------------------------------------------------------------------
__device__ __forceinline__ v16h load_frag(const h16* row, int k0, int h8) {
    const v8h a = *reinterpret_cast<const v8h*>(row + k0 + h8);
    const v8h b = *reinterpret_cast<const v8h*>(row + k0 + 16 + h8);
    v16h r;
#pragma unroll
    for (int i = 0; i < 8; ++i) { r[i] = a[i]; r[i + 8] = b[i]; }
    return r;
}

// ---------------------------------------------------------------------------
// C(M,Nc) = A(M,K)f16 @ Wt(Nc,K)f16^T [+ bias | + bias*rowscale] [relu]
// fp32 accumulate via v_wmma_f32_16x16x32_f16. One 16x16 tile per wave,
// 4 waves per block. Optionally mirrors the fp32 result into fp16.
// ---------------------------------------------------------------------------
template<bool RELU, bool STORE_F16, bool ROWSCALE_BIAS>
__global__ __launch_bounds__(128)
void gemm_wmma(const h16* __restrict__ A, const h16* __restrict__ Bt,
               const float* __restrict__ bias, const float* __restrict__ rowscale,
               float* __restrict__ Cf, h16* __restrict__ Ch,
               int K, int Nc) {
    const int wave = threadIdx.x >> 5;
    const int lane = threadIdx.x & 31;
    const int mt = blockIdx.x;                     // row tile (16 rows)
    const int nt = blockIdx.y * 4 + wave;          // col tile (16 cols)
    const int l15 = lane & 15;
    const int hi  = lane >> 4;
    const int h8  = hi * 8;

    const h16* arow = A  + (size_t)(mt * 16 + l15) * K;
    const h16* brow = Bt + (size_t)(nt * 16 + l15) * K;

    v8f c = {};
    for (int k = 0; k < K; k += 32) {
        v16h a = load_frag(arow, k, h8);
        v16h b = load_frag(brow, k, h8);
        c = __builtin_amdgcn_wmma_f32_16x16x32_f16(
                false, a, false, b, (short)0, c, false, false);
    }

    const int n = nt * 16 + l15;
    const float bn = bias ? bias[n] : 0.0f;
#pragma unroll
    for (int r = 0; r < 8; ++r) {
        const int mm = mt * 16 + r + 8 * hi;       // D layout: VGPR r -> M = r + 8*hi
        float v = c[r];
        if (ROWSCALE_BIAS) v += bn * rowscale[mm]; else v += bn;
        if (RELU) v = fmaxf(v, 0.0f);
        const size_t off = (size_t)mm * Nc + n;
        Cf[off] = v;
        if (STORE_F16) Ch[off] = (h16)v;
    }
}

// Wt[n*K + k] = W[(row_off + k)*N + n]  (fp32 -> fp16 transpose of weights)
__global__ void wt_kernel(const float* __restrict__ W, h16* __restrict__ Wt,
                          int K, int N, int row_off) {
    int idx = blockIdx.x * blockDim.x + threadIdx.x;
    if (idx >= K * N) return;
    int k = idx % K, n = idx / K;
    Wt[(size_t)n * K + k] = (h16)W[(size_t)(row_off + k) * N + n];
}

__global__ void cast_f16_kernel(const float* __restrict__ x, h16* __restrict__ y, int n) {
    int i = blockIdx.x * blockDim.x + threadIdx.x;
    if (i < n) y[i] = (h16)x[i];
}

// s[b,i,h] = sum_j adj[b,i,j] * relu(hi[b,i,h] + hj[b,j,h] + b1[h]);  deg[b,i] = sum_j adj
// One block per (b,i); one channel per thread; adjacency row loads are wave-uniform (SALU).
__global__ __launch_bounds__(GH)
void edge_agg_kernel(const float* __restrict__ hi, const float* __restrict__ hj,
                     const int* __restrict__ adj, const float* __restrict__ b1,
                     float* __restrict__ s, float* __restrict__ deg) {
    const int bi = blockIdx.x;            // b*N + i
    const int b  = bi >> 8;               // N = 256
    const int t  = threadIdx.x;
    const int* arow = adj + (size_t)b * GN * GN + (size_t)(bi & (GN - 1)) * GN;
    const float hv = hi[(size_t)bi * GH + t] + b1[t];
    const float* hjb = hj + (size_t)b * GN * GH;
    float acc = 0.0f, d = 0.0f;
    for (int j = 0; j < GN; ++j) {
        if (arow[j]) {                    // uniform scalar branch
            acc += fmaxf(hv + hjb[(size_t)j * GH + t], 0.0f);
            d += 1.0f;
        }
    }
    s[(size_t)bi * GH + t] = acc;
    if (t == 0) deg[bi] = d;
}

// GRU cell: gi/gh already include their biases (added in GEMM epilogue).
__global__ __launch_bounds__(GH)
void gru_kernel(const float* __restrict__ gi, const float* __restrict__ gh,
                float* __restrict__ h, h16* __restrict__ hh) {
    const int bi = blockIdx.x, t = threadIdx.x;
    const size_t g0 = (size_t)bi * 3 * GH + t;
    const float ir = gi[g0],            hr = gh[g0];
    const float iz = gi[g0 + GH],       hz = gh[g0 + GH];
    const float in = gi[g0 + 2 * GH],   hn = gh[g0 + 2 * GH];
    const float r = 1.0f / (1.0f + __expf(-(ir + hr)));
    const float z = 1.0f / (1.0f + __expf(-(iz + hz)));
    const float n = tanhf(in + r * hn);
    const size_t ho = (size_t)bi * GH + t;
    const float hnew = (1.0f - z) * n + z * h[ho];
    h[ho] = hnew;
    hh[ho] = (h16)hnew;
}

__global__ __launch_bounds__(GH)
void node_sum_kernel(const float* __restrict__ h, float* __restrict__ g) {
    const int b = blockIdx.x, t = threadIdx.x;
    float acc = 0.0f;
    for (int i = 0; i < GN; ++i) acc += h[((size_t)b * GN + i) * GH + t];
    g[(size_t)b * GH + t] = acc;
}

__global__ __launch_bounds__(GH)
void readout1_kernel(const float* __restrict__ g, const float* __restrict__ W1,
                     const float* __restrict__ b1, float* __restrict__ r1) {
    const int b = blockIdx.x, t = threadIdx.x;
    float acc = b1[t];
    for (int k = 0; k < GH; ++k) acc += g[(size_t)b * GH + k] * W1[(size_t)k * GH + t];
    r1[(size_t)b * GH + t] = fmaxf(acc, 0.0f);
}

__global__ void readout2_kernel(const float* __restrict__ r1, const float* __restrict__ W2,
                                const float* __restrict__ b2, float* __restrict__ q) {
    const int idx = threadIdx.x;
    if (idx >= GB * GA) return;
    const int b = idx / GA, a = idx % GA;
    float acc = b2[a];
    for (int k = 0; k < GH; ++k) acc += r1[(size_t)b * GH + k] * W2[(size_t)k * GA + a];
    q[idx] = acc;
}

extern "C" void kernel_launch(void* const* d_in, const int* in_sizes, int n_in,
                              void* d_out, int out_size, void* d_ws, size_t ws_size,
                              hipStream_t stream) {
    const float* node_x  = (const float*)d_in[0];
    const int*   adj     = (const int*)  d_in[1];
    const float* pre_W1  = (const float*)d_in[2];
    const float* pre_b1  = (const float*)d_in[3];
    const float* pre_W2  = (const float*)d_in[4];
    const float* pre_b2  = (const float*)d_in[5];
    const float* msg_W1  = (const float*)d_in[6];
    const float* msg_b1  = (const float*)d_in[7];
    const float* msg_W2  = (const float*)d_in[8];
    const float* msg_b2  = (const float*)d_in[9];
    const float* gru_Wih = (const float*)d_in[10];
    const float* gru_Whh = (const float*)d_in[11];
    const float* gru_bih = (const float*)d_in[12];
    const float* gru_bhh = (const float*)d_in[13];
    const float* ro_W1   = (const float*)d_in[14];
    const float* ro_b1   = (const float*)d_in[15];
    const float* ro_W2   = (const float*)d_in[16];
    const float* ro_b2   = (const float*)d_in[17];
    float* out = (float*)d_out;

    // ---- workspace carve-up -------------------------------------------------
    char* ws = (char*)d_ws;
    size_t o = 0;
    auto alloc = [&](size_t bytes) -> char* {
        char* p = ws + o; o += (bytes + 255) & ~(size_t)255; return p;
    };
    float* h_f32   = (float*)alloc((size_t)GM * GH * 4);
    h16*   h_h     = (h16*)  alloc((size_t)GM * GH * 2);
    h16*   x_h     = (h16*)  alloc((size_t)GM * GF * 2);
    float* t1_f32  = (float*)alloc((size_t)GM * GH * 4);
    h16*   t1_h    = (h16*)  alloc((size_t)GM * GH * 2);
    float* hi_f32  = (float*)alloc((size_t)GM * GH * 4);
    float* hj_f32  = (float*)alloc((size_t)GM * GH * 4);
    float* s_f32   = (float*)alloc((size_t)GM * GH * 4);
    h16*   s_h     = (h16*)  alloc((size_t)GM * GH * 2);
    float* agg_f32 = (float*)alloc((size_t)GM * GH * 4);
    h16*   agg_h   = (h16*)  alloc((size_t)GM * GH * 2);
    float* gi_f32  = (float*)alloc((size_t)GM * 3 * GH * 4);
    float* gh_f32  = (float*)alloc((size_t)GM * 3 * GH * 4);
    float* deg     = (float*)alloc((size_t)GM * 4);
    float* g_f32   = (float*)alloc((size_t)GB * GH * 4);
    float* r1_f32  = (float*)alloc((size_t)GB * GH * 4);
    h16* preW1t = (h16*)alloc((size_t)GH * GF * 2);       // (256,64)
    h16* preW2t = (h16*)alloc((size_t)GH * GH * 2);
    h16* W1it   = (h16*)alloc((size_t)GH * GH * 2);
    h16* W1jt   = (h16*)alloc((size_t)GH * GH * 2);
    h16* msgW2t = (h16*)alloc((size_t)GH * GH * 2);
    h16* Wiht   = (h16*)alloc((size_t)3 * GH * GH * 2);   // (768,256)
    h16* Whht   = (h16*)alloc((size_t)3 * GH * GH * 2);

    // ---- one-time weight transposes to f16 (K-contiguous per output col) ----
    auto wt = [&](const float* W, h16* Wt, int K, int N, int roff) {
        int tot = K * N;
        wt_kernel<<<(tot + 255) / 256, 256, 0, stream>>>(W, Wt, K, N, roff);
    };
    wt(pre_W1, preW1t, GF, GH, 0);
    wt(pre_W2, preW2t, GH, GH, 0);
    wt(msg_W1, W1it,   GH, GH, 0);
    wt(msg_W1, W1jt,   GH, GH, GH);        // rows H..2H-1
    wt(msg_W2, msgW2t, GH, GH, 0);
    wt(gru_Wih, Wiht,  GH, 3 * GH, 0);
    wt(gru_Whh, Whht,  GH, 3 * GH, 0);
    cast_f16_kernel<<<(GM * GF + 255) / 256, 256, 0, stream>>>(node_x, x_h, GM * GF);

    const dim3 blk(128);
    auto grid = [](int Nc) { return dim3(GM / 16, Nc / 64); };

    // ---- pre-net: h = relu(X @ W1 + b1) @ W2 + b2 ---------------------------
    gemm_wmma<true,  true,  false><<<grid(GH), blk, 0, stream>>>(
        x_h, preW1t, pre_b1, nullptr, t1_f32, t1_h, GF, GH);
    gemm_wmma<false, true,  false><<<grid(GH), blk, 0, stream>>>(
        t1_h, preW2t, pre_b2, nullptr, h_f32, h_h, GH, GH);

    // ---- T message-passing iterations --------------------------------------
    for (int t = 0; t < GT; ++t) {
        gemm_wmma<false, false, false><<<grid(GH), blk, 0, stream>>>(
            h_h, W1it, nullptr, nullptr, hi_f32, nullptr, GH, GH);
        gemm_wmma<false, false, false><<<grid(GH), blk, 0, stream>>>(
            h_h, W1jt, nullptr, nullptr, hj_f32, nullptr, GH, GH);
        edge_agg_kernel<<<GM, GH, 0, stream>>>(hi_f32, hj_f32, adj, msg_b1, s_f32, deg);
        cast_f16_kernel<<<(GM * GH + 255) / 256, 256, 0, stream>>>(s_f32, s_h, GM * GH);
        // agg = s @ W2 + b2 * deg   (linearity of W2 pulled through the j-sum)
        gemm_wmma<false, true, true><<<grid(GH), blk, 0, stream>>>(
            s_h, msgW2t, msg_b2, deg, agg_f32, agg_h, GH, GH);
        gemm_wmma<false, false, false><<<grid(3 * GH), blk, 0, stream>>>(
            agg_h, Wiht, gru_bih, nullptr, gi_f32, nullptr, GH, 3 * GH);
        gemm_wmma<false, false, false><<<grid(3 * GH), blk, 0, stream>>>(
            h_h, Whht, gru_bhh, nullptr, gh_f32, nullptr, GH, 3 * GH);
        gru_kernel<<<GM, GH, 0, stream>>>(gi_f32, gh_f32, h_f32, h_h);
    }

    // ---- readout ------------------------------------------------------------
    node_sum_kernel<<<GB, GH, 0, stream>>>(h_f32, g_f32);
    readout1_kernel<<<GB, GH, 0, stream>>>(g_f32, ro_W1, ro_b1, r1_f32);
    readout2_kernel<<<1, 64, 0, stream>>>(r1_f32, ro_W2, ro_b2, out);
}
